// BERTAttention_73959336837413
// MI455X (gfx1250) — matrix operand order repaired and checked by reading, not compiled, not run on hardware
//
#include <hip/hip_runtime.h>
#include <hip/hip_bf16.h>

#define HEADS  12
#define DMODEL 768
#define DHEAD  64
#define SEQ    2048

typedef __attribute__((ext_vector_type(16))) __bf16 v16bf;
typedef __attribute__((ext_vector_type(8)))  float  v8f;
typedef __attribute__((ext_vector_type(4)))  unsigned int v4u;
typedef __attribute__((ext_vector_type(8)))  int v8i;
typedef __attribute__((ext_vector_type(4)))  int v4i;

union FragU { unsigned int u[8]; v16bf b; };
union BF16U { __bf16 b; unsigned short u; };

__device__ inline v8f wmma_bf16(v16bf a, v16bf b, v8f c) {
  return __builtin_amdgcn_wmma_f32_16x16x32_bf16(false, a, false, b, (short)0, c,
                                                 false, false);
}

// A-fragment: 16x32 bf16, row-major source, ISA layout:
// lane<16: M=lane, K groups {0..7,16..23}; lane>=16: M=lane-16, K {8..15,24..31}
__device__ inline v16bf load_frag_a_g(const __bf16* base, int ld, int row0, int col0) {
  const int lane = threadIdx.x & 31;
  const int lr = lane & 15, hf = lane >> 4;
  const unsigned int* p  = (const unsigned int*)(base + (size_t)(row0 + lr) * ld + col0 + hf * 8);
  const unsigned int* p2 = (const unsigned int*)(base + (size_t)(row0 + lr) * ld + col0 + 16 + hf * 8);
  FragU f;
#pragma unroll
  for (int v = 0; v < 4; ++v) f.u[v] = p[v];
#pragma unroll
  for (int v = 0; v < 4; ++v) f.u[4 + v] = p2[v];
  return f.b;
}

// B-fragment: 32x16 bf16 where B[k][n] = src[(n0+n)*ld + k0 + k]
// lane = column n; lane<16: K {0..7,16..23}; lane>=16: K {8..15,24..31}.
__device__ inline v16bf load_frag_b_g(const __bf16* base, int ld, int n0, int k0) {
  const int lane = threadIdx.x & 31;
  const int lr = lane & 15, hf = lane >> 4;
  const unsigned int* p  = (const unsigned int*)(base + (size_t)(n0 + lr) * ld + k0 + hf * 8);
  const unsigned int* p2 = (const unsigned int*)(base + (size_t)(n0 + lr) * ld + k0 + 16 + hf * 8);
  FragU f;
#pragma unroll
  for (int v = 0; v < 4; ++v) f.u[v] = p[v];
#pragma unroll
  for (int v = 0; v < 4; ++v) f.u[4 + v] = p2[v];
  return f.b;
}

// ---- Tensor Data Mover: 2D tile (bf16) global -> LDS, D# per ISA ch.8 ----
__device__ inline void tdm_load_2d_bf16(unsigned lds_off, const void* gtile,
                                        unsigned tile_x, unsigned tile_y,
                                        unsigned long long stride_elems,
                                        unsigned tensor_x, unsigned tensor_y) {
#if __has_builtin(__builtin_amdgcn_tensor_load_to_lds)
  unsigned long long ga = (unsigned long long)(size_t)gtile;
  v4u g0;
  g0[0] = 1u;                                            // count=1 (valid), user mode
  g0[1] = lds_off;                                       // lds_addr (bytes)
  g0[2] = (unsigned)(ga & 0xFFFFFFFFu);                  // global_addr[31:0]
  g0[3] = (unsigned)((ga >> 32) & 0x01FFFFFFu) | (2u << 30);  // addr[56:32] | type=2
  v8i g1;
  g1[0] = (int)(1u << 16);                               // data_size=1 (2 bytes), mask=0
  g1[1] = (int)((tensor_x & 0xFFFFu) << 16);             // tensor_dim0[15:0] @ bits63:48
  g1[2] = (int)(((tensor_x >> 16) & 0xFFFFu) | ((tensor_y & 0xFFFFu) << 16));
  g1[3] = (int)(((tensor_y >> 16) & 0xFFFFu) | ((tile_x & 0xFFFFu) << 16));  // tile_dim0
  g1[4] = (int)(tile_y & 0xFFFFu);                       // tile_dim1 (tile_dim2=0)
  g1[5] = (int)(unsigned)(stride_elems & 0xFFFFFFFFu);   // tensor_dim0_stride[31:0]
  g1[6] = (int)(unsigned)((stride_elems >> 32) & 0xFFFFu);
  g1[7] = 0;
  v4i z4 = {0, 0, 0, 0};
  v8i z8 = {0, 0, 0, 0, 0, 0, 0, 0};
  __builtin_amdgcn_tensor_load_to_lds(g0, g1, z4, z4, z8, 0);
#else
  const __bf16* src = (const __bf16*)gtile;
  (void)src; (void)lds_off; (void)tile_x; (void)tile_y;
  (void)stride_elems; (void)tensor_x; (void)tensor_y;
#endif
}

__global__ void f2bf_kernel(const float* __restrict__ in, __bf16* __restrict__ out, int count) {
  int i = blockIdx.x * blockDim.x + threadIdx.x;
  int stride = gridDim.x * blockDim.x;
  for (; i < count; i += stride) out[i] = (__bf16)in[i];
}

// C = A[M,K] @ W[N,K]^T via bf16 WMMA, f32 accumulate. 32x64 tile per wave.
// mode 0: bf16 row-major [M,N]; mode 1: bf16 transposed [N,ldT]; mode 2: f32 + resid
__global__ __launch_bounds__(128) void gemm_bf16_wmma(
    const __bf16* __restrict__ A, const __bf16* __restrict__ W,
    int M, int N, int K, int mode,
    __bf16* __restrict__ outB, int ldT,
    float* __restrict__ outF, const float* __restrict__ resid) {
  const int wid = threadIdx.x >> 5;
  const int lane = threadIdx.x & 31;
  const int lr = lane & 15, hf = lane >> 4;
  const int row0 = blockIdx.x * 32;
  const int col0 = blockIdx.y * 256 + wid * 64;
  if (row0 >= M || col0 >= N) return;

  v8f acc[2][4] = {};
  for (int k = 0; k < K; k += 32) {
    v16bf a0 = load_frag_a_g(A, K, row0, k);
    v16bf a1 = load_frag_a_g(A, K, row0 + 16, k);
#pragma unroll
    for (int j = 0; j < 4; ++j) {
      v16bf b = load_frag_b_g(W, K, col0 + j * 16, k);
      acc[0][j] = wmma_bf16(a0, b, acc[0][j]);
      acc[1][j] = wmma_bf16(a1, b, acc[1][j]);
    }
  }

#pragma unroll
  for (int i = 0; i < 2; ++i) {
    const int rbase = row0 + i * 16;
    if (mode == 0) {
#pragma unroll
      for (int j = 0; j < 4; ++j)
#pragma unroll
        for (int v = 0; v < 8; ++v) {
          int m = v + hf * 8;
          outB[(size_t)(rbase + m) * N + col0 + j * 16 + lr] = (__bf16)acc[i][j][v];
        }
    } else if (mode == 1) {
#pragma unroll
      for (int j = 0; j < 4; ++j) {
        unsigned int d[4];
#pragma unroll
        for (int v = 0; v < 4; ++v) {
          BF16U lo, hi;
          lo.b = (__bf16)acc[i][j][2 * v];
          hi.b = (__bf16)acc[i][j][2 * v + 1];
          d[v] = (unsigned int)lo.u | ((unsigned int)hi.u << 16);
        }
        unsigned int* q = (unsigned int*)(outB + (size_t)(col0 + j * 16 + lr) * ldT + rbase + hf * 8);
#pragma unroll
        for (int v = 0; v < 4; ++v) q[v] = d[v];
      }
    } else {
#pragma unroll
      for (int j = 0; j < 4; ++j)
#pragma unroll
        for (int v = 0; v < 8; ++v) {
          int m = v + hf * 8;
          size_t idx = (size_t)(rbase + m) * N + col0 + j * 16 + lr;
          outF[idx] = acc[i][j][v] + resid[idx];
        }
    }
  }
}

// Flash attention: one block = one head x 4 consecutive 16-row query tiles.
// K chunk (32 keys x 64 feat) and V chunk (64 feat x 32 keys, from V^T) are
// staged into LDS once per block by the Tensor Data Mover and shared by all
// 4 waves. Q,K bf16 [SEQ,DMODEL]; Vt bf16 [DMODEL,SEQ]; O bf16 [SEQ,DMODEL].
__global__ __launch_bounds__(128) void attn_wmma(
    const __bf16* __restrict__ Q, const __bf16* __restrict__ Km,
    const __bf16* __restrict__ Vt, __bf16* __restrict__ O) {
  __shared__ __bf16 kbuf[32 * 64];   // [key in chunk][feature]
  __shared__ __bf16 vbuf[64 * 32];   // [feature][key in chunk]
  __shared__ __bf16 pt[4][16 * 32];  // per-wave P staging
  const int wid = threadIdx.x >> 5;
  const int lane = threadIdx.x & 31;
  const int lr = lane & 15, hf = lane >> 4;
  const int h  = blockIdx.x >> 5;          // 32 query-groups per head
  const int qt = (blockIdx.x & 31) * 4 + wid;
  const int qr = qt * 16;
  const int hc = h * DHEAD;

  v16bf qa0 = load_frag_a_g(Q, DMODEL, qr, hc);
  v16bf qa1 = load_frag_a_g(Q, DMODEL, qr, hc + 32);

  v8f acc[4] = {};
  float rmax[8], rsum[8];
#pragma unroll
  for (int v = 0; v < 8; ++v) { rmax[v] = -1e30f; rsum[v] = 0.f; }

  const unsigned koff = (unsigned)(size_t)(__bf16*)kbuf;
  const unsigned voff = (unsigned)(size_t)(__bf16*)vbuf;

  for (int kc = 0; kc < SEQ; kc += 32) {
    if (wid == 0) {
      // K tile: rows kc..kc+31 of K, cols hc..hc+63 (contig dim = features)
      tdm_load_2d_bf16(koff, Km + (size_t)kc * DMODEL + hc,
                       /*tile_x=*/64, /*tile_y=*/32, /*stride=*/DMODEL,
                       /*tensor_x=*/DMODEL, /*tensor_y=*/SEQ);
      // V tile: rows hc..hc+63 of V^T, cols kc..kc+31 (contig dim = keys)
      tdm_load_2d_bf16(voff, Vt + (size_t)hc * SEQ + kc,
                       /*tile_x=*/32, /*tile_y=*/64, /*stride=*/SEQ,
                       /*tensor_x=*/SEQ, /*tensor_y=*/DMODEL);
#if __has_builtin(__builtin_amdgcn_s_wait_tensorcnt)
      __builtin_amdgcn_s_wait_tensorcnt(0);
#endif
    }
    __syncthreads();

    // S tile (16 queries x 32 keys): B[k][n] = kbuf[n][k]
    v16bf b00 = load_frag_b_g(kbuf, 64, 0,  0);
    v16bf b01 = load_frag_b_g(kbuf, 64, 0,  32);
    v16bf b10 = load_frag_b_g(kbuf, 64, 16, 0);
    v16bf b11 = load_frag_b_g(kbuf, 64, 16, 32);
    v8f s0 = {}; s0 = wmma_bf16(qa0, b00, s0); s0 = wmma_bf16(qa1, b01, s0);
    v8f s1 = {}; s1 = wmma_bf16(qa0, b10, s1); s1 = wmma_bf16(qa1, b11, s1);
    s0 = s0 * 0.125f;   // 1/sqrt(64)
    s1 = s1 * 0.125f;

    // online softmax: lane's VGPR v holds row m = v + hf*8; row spans the
    // 16 lanes of its half -> butterfly over xor masks 1,2,4,8.
#pragma unroll
    for (int v = 0; v < 8; ++v) {
      float cm = fmaxf(s0[v], s1[v]);
#pragma unroll
      for (int m = 1; m < 16; m <<= 1) cm = fmaxf(cm, __shfl_xor(cm, m, 32));
      float nm   = fmaxf(rmax[v], cm);
      float corr = __expf(rmax[v] - nm);
      float p0 = __expf(s0[v] - nm);
      float p1 = __expf(s1[v] - nm);
      float cs = p0 + p1;
#pragma unroll
      for (int m = 1; m < 16; m <<= 1) cs += __shfl_xor(cs, m, 32);
      rsum[v] = rsum[v] * corr + cs;
      rmax[v] = nm;
#pragma unroll
      for (int j = 0; j < 4; ++j) acc[j][v] *= corr;
      int mrow = v + hf * 8;
      pt[wid][mrow * 32 + lr]      = (__bf16)p0;
      pt[wid][mrow * 32 + 16 + lr] = (__bf16)p1;
    }

    // wave-local LDS RAW fence (cross-lane re-layout D-frag -> A-frag)
    __builtin_amdgcn_wave_barrier();
    asm volatile("s_wait_dscnt 0" ::: "memory");
    __builtin_amdgcn_wave_barrier();

    FragU pf;
    {
      const unsigned int* lp  = (const unsigned int*)&pt[wid][lr * 32 + hf * 8];
      const unsigned int* lp2 = (const unsigned int*)&pt[wid][lr * 32 + 16 + hf * 8];
#pragma unroll
      for (int v = 0; v < 4; ++v) pf.u[v] = lp[v];
#pragma unroll
      for (int v = 0; v < 4; ++v) pf.u[4 + v] = lp2[v];
    }

    // O += P[16x32] @ V[32x64]: B[k][n] = vbuf[n][k]
#pragma unroll
    for (int j = 0; j < 4; ++j) {
      v16bf vb = load_frag_b_g(vbuf, 32, j * 16, 0);
      acc[j] = wmma_bf16(pf.b, vb, acc[j]);
    }

    if (kc + 32 < SEQ)
      __builtin_prefetch(Km + (size_t)(kc + 32 + lr) * DMODEL + hc, 0, 1);

    __syncthreads();   // protect LDS tiles before next TDM overwrite
  }

#pragma unroll
  for (int v = 0; v < 8; ++v) {
    float inv = 1.f / rsum[v];
    int mrow = v + hf * 8;
#pragma unroll
    for (int j = 0; j < 4; ++j)
      O[(size_t)(qr + mrow) * DMODEL + hc + j * 16 + lr] = (__bf16)(acc[j][v] * inv);
  }
}

__global__ __launch_bounds__(256) void ln_kernel(const float* __restrict__ y,
                                                 const float* __restrict__ gamma,
                                                 const float* __restrict__ beta,
                                                 float* __restrict__ out) {
  __shared__ float s1[256], s2[256];
  const int r = blockIdx.x, t = threadIdx.x;
  const float* row = y + (size_t)r * DMODEL;
  float a0 = row[t], a1 = row[t + 256], a2 = row[t + 512];
  s1[t] = a0 + a1 + a2;
  s2[t] = a0 * a0 + a1 * a1 + a2 * a2;
  __syncthreads();
  for (int o = 128; o > 0; o >>= 1) {
    if (t < o) { s1[t] += s1[t + o]; s2[t] += s2[t + o]; }
    __syncthreads();
  }
  float mu   = s1[0] * (1.0f / DMODEL);
  float var  = s2[0] * (1.0f / DMODEL) - mu * mu;
  float rstd = rsqrtf(var + 1e-12f);
  float* orow = out + (size_t)r * DMODEL;
  orow[t]       = (a0 - mu) * rstd * gamma[t]       + beta[t];
  orow[t + 256] = (a1 - mu) * rstd * gamma[t + 256] + beta[t + 256];
  orow[t + 512] = (a2 - mu) * rstd * gamma[t + 512] + beta[t + 512];
}

extern "C" void kernel_launch(void* const* d_in, const int* in_sizes, int n_in,
                              void* d_out, int out_size, void* d_ws, size_t ws_size,
                              hipStream_t stream) {
  (void)in_sizes; (void)n_in; (void)out_size; (void)ws_size;
  const float* x     = (const float*)d_in[0];
  const float* Wq    = (const float*)d_in[1];
  const float* Wk    = (const float*)d_in[2];
  const float* Wv    = (const float*)d_in[3];
  const float* Wo    = (const float*)d_in[4];
  const float* gamma = (const float*)d_in[5];
  const float* beta  = (const float*)d_in[6];
  float* out = (float*)d_out;

  char* ws = (char*)d_ws;
  size_t off = 0;
  auto alloc = [&](size_t bytes) -> void* {
    void* p = ws + off;
    off += bytes;
    off = (off + 255) & ~(size_t)255;
    return p;
  };
  const size_t NX = (size_t)SEQ * DMODEL;
  const size_t NW = (size_t)DMODEL * DMODEL;
  __bf16* xb   = (__bf16*)alloc(NX * 2);
  __bf16* wqb  = (__bf16*)alloc(NW * 2);
  __bf16* wkb  = (__bf16*)alloc(NW * 2);
  __bf16* wvb  = (__bf16*)alloc(NW * 2);
  __bf16* wob  = (__bf16*)alloc(NW * 2);
  __bf16* Qb   = (__bf16*)alloc(NX * 2);
  __bf16* Kb   = (__bf16*)alloc(NX * 2);
  __bf16* Vtb  = (__bf16*)alloc(NX * 2);   // [DMODEL][SEQ]
  __bf16* attb = (__bf16*)alloc(NX * 2);
  float*  yf   = (float*)alloc(NX * 4);

  f2bf_kernel<<<1024, 256, 0, stream>>>(x,  xb,  (int)NX);
  f2bf_kernel<<<1024, 256, 0, stream>>>(Wq, wqb, (int)NW);
  f2bf_kernel<<<1024, 256, 0, stream>>>(Wk, wkb, (int)NW);
  f2bf_kernel<<<1024, 256, 0, stream>>>(Wv, wvb, (int)NW);
  f2bf_kernel<<<1024, 256, 0, stream>>>(Wo, wob, (int)NW);

  dim3 gg(SEQ / 32, DMODEL / 256);  // (64, 3), 128 threads = 4 waves
  gemm_bf16_wmma<<<gg, 128, 0, stream>>>(xb, wqb, SEQ, DMODEL, DMODEL, 0, Qb, 0, nullptr, nullptr);
  gemm_bf16_wmma<<<gg, 128, 0, stream>>>(xb, wkb, SEQ, DMODEL, DMODEL, 0, Kb, 0, nullptr, nullptr);
  gemm_bf16_wmma<<<gg, 128, 0, stream>>>(xb, wvb, SEQ, DMODEL, DMODEL, 1, Vtb, SEQ, nullptr, nullptr);

  attn_wmma<<<HEADS * 32, 128, 0, stream>>>(Qb, Kb, Vtb, attb);

  gemm_bf16_wmma<<<gg, 128, 0, stream>>>(attb, wob, SEQ, DMODEL, DMODEL, 2, nullptr, 0, yf, x);

  ln_kernel<<<SEQ, 256, 0, stream>>>(yf, gamma, beta, out);
}
